// WindowAttention_59751585022223
// MI455X (gfx1250) — matrix operand order repaired and checked by reading, not compiled, not run on hardware
//
#include <hip/hip_runtime.h>

// ---------------- problem constants ----------------
#define B_WIN   2048
#define SEQ     49
#define CDIM    512
#define HEADS   16
#define DHEAD   32
#define NWIN    64
#define MROWS   (B_WIN * SEQ)     // 100352 rows, divisible by 128
#define QKVC    (3 * CDIM)        // 1536
#define SCALE_Q 0.17677669529663687f   // 1/sqrt(32)

typedef _Float16 h8   __attribute__((ext_vector_type(8)));
typedef _Float16 v16h __attribute__((ext_vector_type(16)));
typedef float    v8f  __attribute__((ext_vector_type(8)));
typedef unsigned int u32x4 __attribute__((ext_vector_type(4)));
typedef int          i32x4 __attribute__((ext_vector_type(4)));
typedef int          i32x8 __attribute__((ext_vector_type(8)));

union F8 { v8f v; float f[8]; };
union H8 { h8 v; _Float16 f[8]; };

// Load a 16-half WMMA fragment from two contiguous 16-byte chunks.
__device__ __forceinline__ v16h ld16(const _Float16* p0, const _Float16* p1) {
  h8 lo = *(const h8*)p0;
  h8 hi = *(const h8*)p1;
  return __builtin_shufflevector(lo, hi, 0,1,2,3,4,5,6,7,8,9,10,11,12,13,14,15);
}

// ---------------- TDM: DMA a 2-D f16 tile (tile_w x tile_h) into padded LDS ---
// LDS layout produced: rows of tile_w halves padded to tile_w+8 (pad 4 DWORDs
// after every 16 data DWORDs -> requires tile_w == 32).
#if __has_builtin(__builtin_amdgcn_tensor_load_to_lds)
#define HAVE_TDM 1
__device__ __forceinline__ void tdm_load_2d_f16(unsigned lds_addr, const void* gaddr,
                                                unsigned tile_w, unsigned tile_h,
                                                unsigned row_stride_elems) {
  unsigned long long ga = (unsigned long long)(size_t)gaddr;
  u32x4 g0;
  g0[0] = 1u;                                             // count=1 user D#
  g0[1] = lds_addr;                                       // LDS byte address
  g0[2] = (unsigned)(ga & 0xFFFFFFFFu);                   // global_addr[31:0]
  g0[3] = (unsigned)((ga >> 32) & 0x1FFFFFFu) | 0x80000000u; // addr[56:32] | type=2
  i32x8 g1;
  g1[0] = (int)((1u << 16)        // data_size = 2 bytes
              | (1u << 20)        // pad_enable
              | (3u << 22)        // pad_interval: 16 DWORDs
              | (3u << 25));      // pad_amount: 4 DWORDs
  g1[1] = (int)(tile_w << 16);    // tensor_dim0 (low16)
  g1[2] = (int)(tile_h << 16);    // tensor_dim0 hi=0 | tensor_dim1 (low16)
  g1[3] = (int)(tile_w << 16);    // tensor_dim1 hi=0 | tile_dim0
  g1[4] = (int)(tile_h & 0xFFFF); // tile_dim1 | tile_dim2=0
  g1[5] = (int)row_stride_elems;  // tensor_dim0_stride (low32), elems
  g1[6] = 0;                      // stride hi | tensor_dim1_stride lo
  g1[7] = 0;
  i32x4 z4 = {0, 0, 0, 0};                    // groups 2/3 unused (2-D tensor)
  i32x8 z8 = {0, 0, 0, 0, 0, 0, 0, 0};        // trailing group (clang-23 form)
  __builtin_amdgcn_tensor_load_to_lds(g0, g1, z4, z4, z8, 0);
}
#else
#define HAVE_TDM 0
#endif

// ---------------- prep: weight permute/convert + 2x relative-position bias ----
__global__ void prep_kernel(const float* __restrict__ qkv_w, const float* __restrict__ qkv_b,
                            const float* __restrict__ out_w, const int* __restrict__ rp_index,
                            const float* __restrict__ rp_table,
                            _Float16* __restrict__ wqkv, float* __restrict__ bqkv,
                            _Float16* __restrict__ wout, float* __restrict__ bias2)
{
  int idx = blockIdx.x * 256 + threadIdx.x;
  if (idx < QKVC * CDIM) {
    int r = idx >> 9, c = idx & 511;
    int h = r / 96, rem = r % 96, d = rem / 3, s = rem % 3;
    wqkv[(size_t)(s * CDIM + h * DHEAD + d) * CDIM + c] = (_Float16)qkv_w[idx];
  }
  if (idx < QKVC) {
    int h = idx / 96, rem = idx % 96, d = rem / 3, s = rem % 3;
    bqkv[s * CDIM + h * DHEAD + d] = qkv_b[idx];
  }
  if (idx < CDIM * CDIM) wout[idx] = (_Float16)out_w[idx];
  if (idx < HEADS * SEQ * SEQ) {
    int ij = idx % (SEQ * SEQ);
    int h  = idx / (SEQ * SEQ);
    bias2[idx] = 2.0f * rp_table[rp_index[ij] * HEADS + h];  // reference adds bias twice
  }
}

// ---------------- WMMA GEMM: C[M,Nc] = A[M,K] @ W[Nc,K]^T + bias --------------
// Block tile 128x128, 8 waves, double-buffered LDS with register-staged prefetch.
__global__ __launch_bounds__(256) void gemm_wmma_f16(
    const void* __restrict__ Asrc, int aIsF32,
    const _Float16* __restrict__ W, const float* __restrict__ bias,
    void* __restrict__ Cout, int cIsF32, int Nc, int K)
{
  __shared__ _Float16 As[2][128 * 40];   // 128 rows x 32 halves, stride 40
  __shared__ _Float16 Bs[2][128 * 40];   // 128 C-cols x 32 halves

  const int t    = threadIdx.x;
  const int lane = t & 31;
  const int wave = t >> 5;
  const int g    = (lane >> 4) & 1;
  const int wm   = wave & 1;             // 64-row half
  const int wn   = wave >> 1;            // 32-column slice (0..3)

  v8f zero = {};
  v8f acc[4][2];
  #pragma unroll
  for (int mi = 0; mi < 4; ++mi) { acc[mi][0] = zero; acc[mi][1] = zero; }

  const int    rl   = t >> 1;            // staging row 0..127
  const int    cs   = (t & 1) << 4;      // 0 or 16 halves
  const size_t arow = (size_t)blockIdx.x * 128 + rl;
  const size_t brow = (size_t)blockIdx.y * 128 + rl;

  float4 fa0, fa1, fa2, fa3;
  h8 ha0, ha1, hb0, hb1;

  auto fetch = [&](int kt) {
    const int ko = kt * 32 + cs;
    if (aIsF32) {
      const float* Ap = (const float*)Asrc + arow * K + ko;
      fa0 = ((const float4*)Ap)[0];
      fa1 = ((const float4*)Ap)[1];
      fa2 = ((const float4*)Ap)[2];
      fa3 = ((const float4*)Ap)[3];
    } else {
      const _Float16* Ap = (const _Float16*)Asrc + arow * K + ko;
      ha0 = ((const h8*)Ap)[0];
      ha1 = ((const h8*)Ap)[1];
    }
    const _Float16* Wp = W + brow * K + ko;
    hb0 = ((const h8*)Wp)[0];
    hb1 = ((const h8*)Wp)[1];
  };
  auto commit = [&](int bsel) {
    if (aIsF32) {
      H8 u0, u1;
      u0.f[0] = (_Float16)fa0.x; u0.f[1] = (_Float16)fa0.y;
      u0.f[2] = (_Float16)fa0.z; u0.f[3] = (_Float16)fa0.w;
      u0.f[4] = (_Float16)fa1.x; u0.f[5] = (_Float16)fa1.y;
      u0.f[6] = (_Float16)fa1.z; u0.f[7] = (_Float16)fa1.w;
      u1.f[0] = (_Float16)fa2.x; u1.f[1] = (_Float16)fa2.y;
      u1.f[2] = (_Float16)fa2.z; u1.f[3] = (_Float16)fa2.w;
      u1.f[4] = (_Float16)fa3.x; u1.f[5] = (_Float16)fa3.y;
      u1.f[6] = (_Float16)fa3.z; u1.f[7] = (_Float16)fa3.w;
      *(h8*)(&As[bsel][rl * 40 + cs])     = u0.v;
      *(h8*)(&As[bsel][rl * 40 + cs + 8]) = u1.v;
    } else {
      *(h8*)(&As[bsel][rl * 40 + cs])     = ha0;
      *(h8*)(&As[bsel][rl * 40 + cs + 8]) = ha1;
    }
    *(h8*)(&Bs[bsel][rl * 40 + cs])     = hb0;
    *(h8*)(&Bs[bsel][rl * 40 + cs + 8]) = hb1;
  };

  const int kIters = K >> 5;
  fetch(0);
  commit(0);
  __syncthreads();
  for (int kt = 0; kt < kIters; ++kt) {
    const int cur = kt & 1;
    if (kt + 1 < kIters) fetch(kt + 1);     // overlap HBM with WMMA
    const _Float16* bb0 = &Bs[cur][(wn * 32 + (lane & 15)) * 40 + g * 16];
    const _Float16* bb1 = bb0 + 16 * 40;
    v16h bf0 = ld16(bb0, bb0 + 8);
    v16h bf1 = ld16(bb1, bb1 + 8);
    #pragma unroll
    for (int mi = 0; mi < 4; ++mi) {
      const _Float16* ap = &As[cur][(wm * 64 + mi * 16 + (lane & 15)) * 40];
      v16h af = ld16(ap + g * 16, ap + g * 16 + 8);
      acc[mi][0] = __builtin_amdgcn_wmma_f32_16x16x32_f16(
          false, af, false, bf0, (short)0, acc[mi][0], false, false);
      acc[mi][1] = __builtin_amdgcn_wmma_f32_16x16x32_f16(
          false, af, false, bf1, (short)0, acc[mi][1], false, false);
    }
    if (kt + 1 < kIters) commit(cur ^ 1);
    __syncthreads();
  }

  #pragma unroll
  for (int ni = 0; ni < 2; ++ni) {
    const int   col = blockIdx.y * 128 + wn * 32 + ni * 16 + (lane & 15);
    const float bv  = bias[col];
    #pragma unroll
    for (int mi = 0; mi < 4; ++mi) {
      F8 u; u.v = acc[mi][ni];
      #pragma unroll
      for (int r = 0; r < 8; ++r) {
        size_t row = (size_t)blockIdx.x * 128 + wm * 64 + mi * 16 + r + 8 * g;
        float  val = u.f[r] + bv;
        if (cIsF32) ((float*)Cout)[row * Nc + col] = val;
        else        ((_Float16*)Cout)[row * Nc + col] = (_Float16)val;
      }
    }
  }
}

// ---------------- fused per-head attention -----------------------------------
// grid (B_WIN, HEADS), block 32: one wave per (window, head), 40960 B LDS.
// Q/K tiles staged by the Tensor Data Mover; V staged manually (transposed).
__global__ __launch_bounds__(32) void attn_wmma(
    const _Float16* __restrict__ qkv, const float* __restrict__ bias2,
    const float* __restrict__ mask, _Float16* __restrict__ attnout)
{
  extern __shared__ char smem[];
  const int lane = threadIdx.x & 31;
  const int g    = (lane >> 4) & 1;
  const int b    = blockIdx.x;
  const int head = blockIdx.y;          // SGPR -> uniform TDM descriptors

  _Float16* q_s  = (_Float16*)smem;              // 64 x 40 halves
  _Float16* k_s  = q_s + 64 * 40;                // 64 x 40
  _Float16* vt_s = k_s + 64 * 40;                // 32 x 72 (V transposed [d][n])
  float*    S_s  = (float*)(vt_s + 32 * 72);     // 64 x 66 fp32 scores
  _Float16* P_s  = (_Float16*)(S_s + 64 * 66);   // 64 x 72 f16 probabilities

  const _Float16* qg = qkv + (size_t)b * SEQ * QKVC + head * DHEAD;  // Q cols
  const _Float16* kg = qg + CDIM;                                    // K cols
  const _Float16* vg = qg + 2 * CDIM;                                // V cols

#if HAVE_TDM
  // async DMA: 49x32-elem tiles, row stride 1536 elems, HW pad to 40-half rows
  tdm_load_2d_f16((unsigned)(size_t)q_s, qg, 32, SEQ, QKVC);
  tdm_load_2d_f16((unsigned)(size_t)k_s, kg, 32, SEQ, QKVC);
#endif

  // zero exactly the cells the DMA does not write (no DS<->TDM overlap)
  {
    h8 hz = {};
    for (int i = lane; i < 60; i += 32) {          // q/k rows 49..63
      int rr = 49 + (i >> 2), c8 = (i & 3) << 3;
      *(h8*)(q_s + rr * 40 + c8) = hz;
      *(h8*)(k_s + rr * 40 + c8) = hz;
    }
    for (int i = lane; i < 64; i += 32) {          // vt cols 48..63
      int d = i >> 1, c8 = 48 + ((i & 1) << 3);
      *(h8*)(vt_s + d * 72 + c8) = hz;
    }
    for (int i = lane; i < 120; i += 32) {         // P rows 49..63
      int rr = 49 + (i >> 3), c8 = (i & 7) << 3;
      *(h8*)(P_s + rr * 72 + c8) = hz;
    }
  }

#if !HAVE_TDM
  for (int idx = lane; idx < SEQ * 4; idx += 32) { // manual q/k staging fallback
    int n = idx >> 2, c8 = (idx & 3) << 3;
    *(h8*)(q_s + n * 40 + c8) = *(const h8*)(qg + (size_t)n * QKVC + c8);
    *(h8*)(k_s + n * 40 + c8) = *(const h8*)(kg + (size_t)n * QKVC + c8);
  }
#endif

  // stage V transposed (TDM cannot transpose)
  for (int idx = lane; idx < SEQ * 4; idx += 32) {
    int n = idx >> 2, c8 = (idx & 3) << 3;
    H8 u; u.v = *(const h8*)(vg + (size_t)n * QKVC + c8);
    #pragma unroll
    for (int j = 0; j < 8; ++j) vt_s[(c8 + j) * 72 + n] = u.f[j];
  }

#if HAVE_TDM
  __builtin_amdgcn_s_wait_tensorcnt(0);
#endif
  __syncthreads();

  // S = scale * (q @ k^T): 4x4 tiles of 16x16, one K=32 WMMA step each
  v16h kf[4];
  #pragma unroll
  for (int nj = 0; nj < 4; ++nj) {
    const _Float16* kp = k_s + (nj * 16 + (lane & 15)) * 40 + g * 16;
    kf[nj] = ld16(kp, kp + 8);
  }
  #pragma unroll
  for (int mi = 0; mi < 4; ++mi) {
    const _Float16* qp = q_s + (mi * 16 + (lane & 15)) * 40;
    v16h af = ld16(qp + g * 16, qp + g * 16 + 8);
    #pragma unroll
    for (int nj = 0; nj < 4; ++nj) {
      v8f acc = {};
      acc = __builtin_amdgcn_wmma_f32_16x16x32_f16(
          false, af, false, kf[nj], (short)0, acc, false, false);
      F8 u; u.v = acc;
      int col = nj * 16 + (lane & 15);
      #pragma unroll
      for (int r = 0; r < 8; ++r)
        S_s[(mi * 16 + r + 8 * g) * 66 + col] = u.f[r] * SCALE_Q;
    }
  }
  __syncthreads();

  // fp32 softmax over 49 valid columns, with doubled rel-pos bias + window mask
  {
    const float* b2 = bias2 + head * (SEQ * SEQ);
    const float* mk = mask + (size_t)(b & (NWIN - 1)) * SEQ * SEQ;
    for (int i = lane; i < SEQ; i += 32) {
      float*       Sr  = S_s + i * 66;
      const float* b2r = b2 + i * SEQ;
      const float* mkr = mk + i * SEQ;
      float mx = -1e30f;
      for (int j = 0; j < SEQ; ++j) {
        float v = Sr[j] + b2r[j] + mkr[j];
        Sr[j] = v;
        mx = fmaxf(mx, v);
      }
      float sum = 0.f;
      for (int j = 0; j < SEQ; ++j) {
        float e = __expf(Sr[j] - mx);
        Sr[j] = e;
        sum += e;
      }
      float inv = 1.0f / sum;
      _Float16* Pr = P_s + i * 72;
      for (int j = 0; j < SEQ; ++j)   Pr[j] = (_Float16)(Sr[j] * inv);
      for (int j = SEQ; j < 64; ++j)  Pr[j] = (_Float16)0.f;
    }
  }
  __syncthreads();

  // out = P @ V : A = P (64x64, two K=32 steps), B = Vt (contiguous-K per lane)
  v16h vf[2][2];
  #pragma unroll
  for (int dt = 0; dt < 2; ++dt)
    #pragma unroll
    for (int kc = 0; kc < 2; ++kc) {
      const _Float16* vp = vt_s + (dt * 16 + (lane & 15)) * 72 + kc * 32 + g * 16;
      vf[dt][kc] = ld16(vp, vp + 8);
    }
  #pragma unroll
  for (int mi = 0; mi < 4; ++mi) {
    const _Float16* pp = P_s + (mi * 16 + (lane & 15)) * 72;
    v16h a0 = ld16(pp + g * 16,      pp + g * 16 + 8);
    v16h a1 = ld16(pp + 32 + g * 16, pp + 40 + g * 16);
    #pragma unroll
    for (int dt = 0; dt < 2; ++dt) {
      v8f acc = {};
      acc = __builtin_amdgcn_wmma_f32_16x16x32_f16(
          false, a0, false, vf[dt][0], (short)0, acc, false, false);
      acc = __builtin_amdgcn_wmma_f32_16x16x32_f16(
          false, a1, false, vf[dt][1], (short)0, acc, false, false);
      F8 u; u.v = acc;
      #pragma unroll
      for (int r = 0; r < 8; ++r) {
        int row = mi * 16 + r + 8 * g;
        if (row < SEQ)
          attnout[((size_t)b * SEQ + row) * CDIM + head * DHEAD + dt * 16 + (lane & 15)]
              = (_Float16)u.f[r];
      }
    }
  }
}

// ---------------- host-side orchestration ------------------------------------
extern "C" void kernel_launch(void* const* d_in, const int* in_sizes, int n_in,
                              void* d_out, int out_size, void* d_ws, size_t ws_size,
                              hipStream_t stream) {
  (void)in_sizes; (void)n_in; (void)out_size; (void)ws_size;
  const float* x        = (const float*)d_in[0];
  const float* qkv_w    = (const float*)d_in[1];
  const float* qkv_b    = (const float*)d_in[2];
  const int*   rp_index = (const int*)d_in[3];
  const float* rp_table = (const float*)d_in[4];
  const float* out_w    = (const float*)d_in[5];
  const float* out_b    = (const float*)d_in[6];
  const float* mask     = (const float*)d_in[7];

  char*  ws  = (char*)d_ws;
  size_t off = 0;
  auto carve = [&](size_t bytes) -> void* {
    void* p = ws + off;
    off = (off + bytes + 255) & ~(size_t)255;
    return p;
  };
  _Float16* wqkv    = (_Float16*)carve((size_t)QKVC * CDIM * sizeof(_Float16));
  _Float16* wout    = (_Float16*)carve((size_t)CDIM * CDIM * sizeof(_Float16));
  float*    bqkv    = (float*)carve((size_t)QKVC * sizeof(float));
  float*    bias2   = (float*)carve((size_t)HEADS * SEQ * SEQ * sizeof(float));
  _Float16* qkvbuf  = (_Float16*)carve((size_t)MROWS * QKVC * sizeof(_Float16));
  _Float16* attnout = (_Float16*)carve((size_t)MROWS * CDIM * sizeof(_Float16));

  // 1) weight permute/convert + doubled relative-position bias table
  prep_kernel<<<dim3((QKVC * CDIM + 255) / 256), 256, 0, stream>>>(
      qkv_w, qkv_b, out_w, rp_index, rp_table, wqkv, bqkv, wout, bias2);

  // 2) QKV projection: [100352,512] fp32 x [1536,512]^T f16 -> f16 (Q|K|V per row)
  gemm_wmma_f16<<<dim3(MROWS / 128, QKVC / 128), 256, 0, stream>>>(
      x, 1, wqkv, bqkv, qkvbuf, 0, QKVC, CDIM);

  // 3) fused per-head attention (TDM-staged q/k, scale, 2x bias, mask, softmax, PV)
  attn_wmma<<<dim3(B_WIN, HEADS), 32, 40960, stream>>>(
      qkvbuf, bias2, mask, attnout);

  // 4) output projection: [100352,512] f16 x [512,512]^T f16 -> fp32 d_out
  gemm_wmma_f16<<<dim3(MROWS / 128, CDIM / 128), 256, 0, stream>>>(
      attnout, 0, wout, out_b, d_out, 1, CDIM, CDIM);
}
